// Head_50525995270480
// MI455X (gfx1250) — matrix operand order repaired and checked by reading, not compiled, not run on hardware
//
#include <hip/hip_runtime.h>

typedef __attribute__((ext_vector_type(16))) __bf16 v16bf;
typedef __attribute__((ext_vector_type(8)))  __bf16 v8bf;
typedef __attribute__((ext_vector_type(8)))  float  v8f;
typedef __attribute__((ext_vector_type(4)))  float  v4f;

#define T_SEQ 4096
#define C_DIM 1024
#define D_DIM 64

// ---------------------------------------------------------------------------
// Kernel 1: transpose + convert weights to bf16.  Wt[which][d][c].
// Wq gets scale D^-0.5 * log2(e) folded in, so attention works in exp2 domain.
// ---------------------------------------------------------------------------
__global__ __launch_bounds__(256) void prep_w(const float* __restrict__ Wq,
                                              const float* __restrict__ Wk,
                                              const float* __restrict__ Wv,
                                              __bf16* __restrict__ wt) {
    int i = blockIdx.x * blockDim.x + threadIdx.x;     // 0 .. 3*64*1024-1
    if (i >= 3 * D_DIM * C_DIM) return;
    int which = i >> 16;          // 64*1024 = 65536 per matrix
    int rem   = i & 65535;
    int d     = rem >> 10;
    int c     = rem & 1023;
    const float* W = (which == 0) ? Wq : (which == 1) ? Wk : Wv;
    float v = W[c * D_DIM + d];
    if (which == 0) v *= 0.125f * 1.44269504088896f;   // D^-0.5 * log2(e)
    wt[i] = (__bf16)v;                                 // i == which*65536 + d*1024 + c
}

// ---------------------------------------------------------------------------
// Kernel 2: QKV projection.  Block = 8 waves = 32 rows x 64 cols, 3 outputs.
// A-frag: x f32 -> bf16 in-register.  B-frag: Wt contiguous 32B per lane.
// Q,K stored row-major bf16 [B*T][64]; V stored transposed Vt[b][64][T].
// ---------------------------------------------------------------------------
__global__ __launch_bounds__(256) void qkv_proj(const float* __restrict__ x,
                                                const __bf16* __restrict__ wt,
                                                __bf16* __restrict__ qb,
                                                __bf16* __restrict__ kbuf,
                                                __bf16* __restrict__ vt) {
    const int lane = threadIdx.x & 31;
    const int w    = threadIdx.x >> 5;
    const int mt   = w >> 2;                 // 0..1  (16-row tile within block)
    const int nt   = w & 3;                  // 0..3  (16-col tile of D=64)
    const int ln   = lane & 15;
    const int hi   = lane >> 4;

    const int M0   = blockIdx.x * 32 + mt * 16;      // global row (B*T flat)
    const int rowA = M0 + ln;
    const int rowB = nt * 16 + ln;

    v8f acc0 = {}; v8f acc1 = {}; v8f acc2 = {};

    for (int ks = 0; ks < C_DIM / 32; ++ks) {
        // ---- A fragment: 16x32 bf16 from f32 x -------------------------------
        const float* px0 = x + rowA * C_DIM + ks * 32 + hi * 16;  // K 0..7 / 16..23
        const float* px1 = px0 + 8;                                // K 8..15 / 24..31
        v4f a0 = *(const v4f*)(px0);
        v4f a1 = *(const v4f*)(px0 + 4);
        v4f a2 = *(const v4f*)(px1);
        v4f a3 = *(const v4f*)(px1 + 4);
        v16bf A;
#pragma unroll
        for (int e = 0; e < 4; ++e) {
            A[e]      = (__bf16)a0[e];
            A[4 + e]  = (__bf16)a1[e];
            A[8 + e]  = (__bf16)a2[e];
            A[12 + e] = (__bf16)a3[e];
        }
        // ---- B fragments: 32x16 bf16, contiguous 32B per lane ----------------
        const __bf16* pw = wt + rowB * C_DIM + ks * 32 + hi * 16;
        v16bf B0 = *(const v16bf*)(pw);                 // Wq^T (pre-scaled)
        v16bf B1 = *(const v16bf*)(pw + 65536);         // Wk^T
        v16bf B2 = *(const v16bf*)(pw + 131072);        // Wv^T

        acc0 = __builtin_amdgcn_wmma_f32_16x16x32_bf16(false, A, false, B0,
                                                       (short)0, acc0, false, false);
        acc1 = __builtin_amdgcn_wmma_f32_16x16x32_bf16(false, A, false, B1,
                                                       (short)0, acc1, false, false);
        acc2 = __builtin_amdgcn_wmma_f32_16x16x32_bf16(false, A, false, B2,
                                                       (short)0, acc2, false, false);
    }

    const int colN = nt * 16 + ln;
    // Q, K: row-major stores (C layout: VGPR r -> row M0 + r + 8*hi, lane -> col)
#pragma unroll
    for (int r = 0; r < 8; ++r) {
        int row = M0 + r + 8 * hi;
        qb[row * D_DIM + colN]   = (__bf16)acc0[r];
        kbuf[row * D_DIM + colN] = (__bf16)acc1[r];
    }
    // V: transposed store -> Vt[b][d][t]; 8 consecutive t per lane = one 16B store
    const int b  = M0 >> 12;                     // /4096
    const int tl = (M0 & 4095) + 8 * hi;
    v8bf vv;
#pragma unroll
    for (int r = 0; r < 8; ++r) vv[r] = (__bf16)acc2[r];
    *(v8bf*)(vt + (size_t)b * D_DIM * T_SEQ + (size_t)colN * T_SEQ + tl) = vv;
}

// ---------------------------------------------------------------------------
// Kernel 3: causal flash attention.  One wave per 16-query tile, 64 keys/iter.
// 2 waves/block for fine-grained scheduling across WGPs.
// ---------------------------------------------------------------------------
__global__ __launch_bounds__(64) void attn(const __bf16* __restrict__ qb,
                                           const __bf16* __restrict__ kbuf,
                                           const __bf16* __restrict__ vt,
                                           float* __restrict__ out) {
    __shared__ __align__(16) __bf16 plds[2][16][64];   // per-wave P transpose scratch

    const int lane = threadIdx.x & 31;
    const int w    = threadIdx.x >> 5;
    const int ln   = lane & 15;
    const int hi   = lane >> 4;

    const int qt = blockIdx.x * 2 + w;                 // 0 .. 511
    const int b  = qt >> 8;                            // / (T/16)
    const int t0 = (qt & 255) << 4;

    const __bf16* Q = qb   + (size_t)b * T_SEQ * D_DIM;
    const __bf16* K = kbuf + (size_t)b * T_SEQ * D_DIM;
    const __bf16* V = vt   + (size_t)b * D_DIM * T_SEQ;
    float*        O = out  + (size_t)b * T_SEQ * D_DIM;

    // Q A-fragments for both 32-wide K steps of the d=64 reduction
    v16bf qf[2];
#pragma unroll
    for (int ks = 0; ks < 2; ++ks) {
        const __bf16* p0 = Q + (t0 + ln) * D_DIM + ks * 32 + hi * 16;
        v8bf a0 = *(const v8bf*)(p0);
        v8bf a1 = *(const v8bf*)(p0 + 8);
#pragma unroll
        for (int e = 0; e < 8; ++e) { qf[ks][e] = a0[e]; qf[ks][8 + e] = a1[e]; }
    }

    float m[8], lsum[8];
#pragma unroll
    for (int r = 0; r < 8; ++r) { m[r] = -1e30f; lsum[r] = 0.0f; }
    v8f o[4] = {{}, {}, {}, {}};

    const int nj = (t0 + 79) >> 6;                     // # of 64-key blocks (causal)
    for (int j = 0; j < nj; ++j) {
        const int kb0 = j * 64;

        if (j + 1 < nj) {                              // gfx1250 global_prefetch_b8
            __builtin_prefetch(K + (kb0 + 64 + ln) * D_DIM, 0, 1);
            __builtin_prefetch(K + (kb0 + 96 + ln) * D_DIM, 0, 1);
            __builtin_prefetch(V + (size_t)ln * T_SEQ + kb0 + 64, 0, 1);
        }

        // ---- S = Q K^T : four 16x16 key tiles (64 keys) ----------------------
        v8f s[4] = {{}, {}, {}, {}};
#pragma unroll
        for (int ks = 0; ks < 2; ++ks) {
            const __bf16* pk = K + (kb0 + ln) * D_DIM + ks * 32 + hi * 16;
#pragma unroll
            for (int t = 0; t < 4; ++t) {
                v16bf bk = *(const v16bf*)(pk + t * 16 * D_DIM);
                s[t] = __builtin_amdgcn_wmma_f32_16x16x32_bf16(false, qf[ks], false, bk,
                                                               (short)0, s[t], false, false);
            }
        }

        // ---- causal mask (only near the diagonal) ----------------------------
        if (kb0 + 63 > t0) {
#pragma unroll
            for (int t = 0; t < 4; ++t) {
                int key = kb0 + t * 16 + ln;
#pragma unroll
                for (int r = 0; r < 8; ++r) {
                    if (key > t0 + r + 8 * hi) s[t][r] = -1e30f;
                }
            }
        }

        // ---- online softmax in exp2 domain.  Row max must be lane-uniform
        //      (4 shuffles); row sum kept as per-lane partial (reduced at end).
#pragma unroll
        for (int r = 0; r < 8; ++r) {
            float mv = fmaxf(fmaxf(s[0][r], s[1][r]), fmaxf(s[2][r], s[3][r]));
            mv = fmaxf(mv, __shfl_xor(mv, 8));
            mv = fmaxf(mv, __shfl_xor(mv, 4));
            mv = fmaxf(mv, __shfl_xor(mv, 2));
            mv = fmaxf(mv, __shfl_xor(mv, 1));
            float mn = fmaxf(m[r], mv);
            float al = exp2f(m[r] - mn);
            m[r] = mn;
            float e0 = exp2f(s[0][r] - mn);
            float e1 = exp2f(s[1][r] - mn);
            float e2 = exp2f(s[2][r] - mn);
            float e3 = exp2f(s[3][r] - mn);
            lsum[r] = lsum[r] * al + ((e0 + e1) + (e2 + e3));
            o[0][r] *= al; o[1][r] *= al; o[2][r] *= al; o[3][r] *= al;
            int row = r + 8 * hi;                       // C-layout -> LDS [row][key]
            plds[w][row][ln]      = (__bf16)e0;
            plds[w][row][ln + 16] = (__bf16)e1;
            plds[w][row][ln + 32] = (__bf16)e2;
            plds[w][row][ln + 48] = (__bf16)e3;
        }

        // ---- P A-fragments (16x32 each) back from LDS ------------------------
        v16bf pf0, pf1;
        {
            v8bf a0 = *(const v8bf*)&plds[w][ln][hi * 16];
            v8bf a1 = *(const v8bf*)&plds[w][ln][8 + hi * 16];
            v8bf a2 = *(const v8bf*)&plds[w][ln][32 + hi * 16];
            v8bf a3 = *(const v8bf*)&plds[w][ln][40 + hi * 16];
#pragma unroll
            for (int e = 0; e < 8; ++e) {
                pf0[e] = a0[e]; pf0[8 + e] = a1[e];
                pf1[e] = a2[e]; pf1[8 + e] = a3[e];
            }
        }

        // ---- O += P V : V^T stored, so B-frags are contiguous 32B ------------
        const __bf16* pv = V + (size_t)ln * T_SEQ + kb0 + hi * 16;
#pragma unroll
        for (int nt = 0; nt < 4; ++nt) {
            v16bf bv0 = *(const v16bf*)(pv + (size_t)nt * 16 * T_SEQ);       // keys 0..31
            v16bf bv1 = *(const v16bf*)(pv + (size_t)nt * 16 * T_SEQ + 32);  // keys 32..63
            o[nt] = __builtin_amdgcn_wmma_f32_16x16x32_bf16(false, pf0, false, bv0,
                                                            (short)0, o[nt], false, false);
            o[nt] = __builtin_amdgcn_wmma_f32_16x16x32_bf16(false, pf1, false, bv1,
                                                            (short)0, o[nt], false, false);
        }
    }

    // ---- epilogue: finish deferred row-sum reduction, normalize, store -------
#pragma unroll
    for (int r = 0; r < 8; ++r) {
        float rs = lsum[r];
        rs += __shfl_xor(rs, 8);
        rs += __shfl_xor(rs, 4);
        rs += __shfl_xor(rs, 2);
        rs += __shfl_xor(rs, 1);
        float inv = 1.0f / rs;
        int row = t0 + r + 8 * hi;
        O[row * D_DIM + 0  + ln] = o[0][r] * inv;
        O[row * D_DIM + 16 + ln] = o[1][r] * inv;
        O[row * D_DIM + 32 + ln] = o[2][r] * inv;
        O[row * D_DIM + 48 + ln] = o[3][r] * inv;
    }
}

// ---------------------------------------------------------------------------
extern "C" void kernel_launch(void* const* d_in, const int* in_sizes, int n_in,
                              void* d_out, int out_size, void* d_ws, size_t ws_size,
                              hipStream_t stream) {
    (void)in_sizes; (void)n_in; (void)out_size; (void)ws_size;
    const float* x  = (const float*)d_in[0];
    const float* Wq = (const float*)d_in[1];
    const float* Wk = (const float*)d_in[2];
    const float* Wv = (const float*)d_in[3];
    float* out = (float*)d_out;

    char* ws = (char*)d_ws;
    __bf16* wt = (__bf16*)(ws);                                  // 3*64*1024*2   = 393216 B
    __bf16* qb = (__bf16*)(ws + 393216);                         // 8192*64*2     = 1 MiB
    __bf16* kb = (__bf16*)(ws + 393216 + 1048576);
    __bf16* vt = (__bf16*)(ws + 393216 + 2 * 1048576);           // total ~3.4 MiB

    prep_w  <<<(3 * 64 * 1024 + 255) / 256, 256, 0, stream>>>(Wq, Wk, Wv, wt);
    qkv_proj<<<(2 * T_SEQ) / 32,            256, 0, stream>>>(x, wt, qb, kb, vt);
    attn    <<<(2 * T_SEQ / 16) / 2,        64,  0, stream>>>(qb, kb, vt, out);
}
